// MultiHeadSelfAttention_55748675502749
// MI455X (gfx1250) — matrix operand order repaired and checked by reading, not compiled, not run on hardware
//
#include <hip/hip_runtime.h>

// ---------------------------------------------------------------------------
// MI455X (gfx1250) multi-head self-attention, bf16 WMMA + flash attention.
// B=4, C=2048, D=1024, H=16, Dh=64.  Wave32, v_wmma_f32_16x16x32_bf16,
// double-buffered async global->LDS staging (ASYNCcnt pipeline).
// ---------------------------------------------------------------------------

typedef __attribute__((ext_vector_type(16))) __bf16 v16bf;
typedef __attribute__((ext_vector_type(8)))  __bf16 v8bf;
typedef __attribute__((ext_vector_type(4)))  __bf16 v4bf;
typedef __attribute__((ext_vector_type(2)))  __bf16 v2bf;
typedef __attribute__((ext_vector_type(8)))  float  v8f;
typedef __attribute__((ext_vector_type(4)))  int    v4i;

#define WMMA_BF16(a, b, c) \
  __builtin_amdgcn_wmma_f32_16x16x32_bf16(false, (a), false, (b), (short)0, (c), false, false)

#define AS1 __attribute__((address_space(1)))
#define AS3 __attribute__((address_space(3)))

#if defined(__HIP_DEVICE_COMPILE__) && __has_builtin(__builtin_amdgcn_global_load_async_to_lds_b128) && __has_builtin(__builtin_amdgcn_s_wait_asynccnt)
#define USE_ASYNC 1
#else
#define USE_ASYNC 0
#endif

// 16B global -> LDS copy: async (no VGPR round-trip, ASYNCcnt) when available.
static __device__ __forceinline__ void cp16(const void* g, void* l) {
#if USE_ASYNC
  __builtin_amdgcn_global_load_async_to_lds_b128((AS1 v4i*)g, (AS3 v4i*)l, 0, 0);
#else
  *(v8bf*)l = *(const v8bf*)g;
#endif
}
static __device__ __forceinline__ void cp_wait() {
#if USE_ASYNC
  __builtin_amdgcn_s_wait_asynccnt(0);
#endif
}

// A-fragment (16x32 bf16, M=lane%16): lane<16 holds K 0..15,
// lane>=16 holds K 16..31 -> two 16B chunks at p+2*half, p+2*half+8.
static __device__ __forceinline__ v16bf ld_fragA(const __bf16* p, int half) {
  v8bf lo = *(const v8bf*)(p + 2 * half);
  v8bf hi = *(const v8bf*)(p + 2 * half + 8);
  return __builtin_shufflevector(lo, hi, 0,1,2,3,4,5,6,7,8,9,10,11,12,13,14,15);
}

// B-fragment (32x16 bf16, N=lane%16): lane<16 holds K 0..15, lane>=16 holds
// K 16..31 -> one contiguous 32B run at p+koff (koff = 0 or 16).
static __device__ __forceinline__ v16bf ld_fragB(const __bf16* p, int koff) {
  v8bf lo = *(const v8bf*)(p + koff);
  v8bf hi = *(const v8bf*)(p + koff + 8);
  return __builtin_shufflevector(lo, hi, 0,1,2,3,4,5,6,7,8,9,10,11,12,13,14,15);
}

// ---------------------------------------------------------------------------
// Precision prep: f32 -> bf16 elementwise, and f32 [K][N] -> bf16 [N][K].
// ---------------------------------------------------------------------------
__global__ __launch_bounds__(256)
void cvt_f32_bf16(const float* __restrict__ src, __bf16* __restrict__ dst, int n4) {
  const int i = blockIdx.x * 256 + threadIdx.x;
  if (i < n4) {
    const float4 f = ((const float4*)src)[i];
    v4bf o;
    o[0] = (__bf16)f.x; o[1] = (__bf16)f.y; o[2] = (__bf16)f.z; o[3] = (__bf16)f.w;
    ((v4bf*)dst)[i] = o;
  }
}

__global__ __launch_bounds__(256)
void transpose_cvt(const float* __restrict__ W, __bf16* __restrict__ Wt,
                   int K, int N) {
  __shared__ __bf16 sT[32][33];
  const int n0 = blockIdx.x * 32, k0 = blockIdx.y * 32;
  const int r = threadIdx.x >> 5, c = threadIdx.x & 31;
#pragma unroll
  for (int p = 0; p < 4; ++p) {
    const int k = p * 8 + r;
    sT[c][k] = (__bf16)W[(size_t)(k0 + k) * N + n0 + c];
  }
  __syncthreads();
#pragma unroll
  for (int p = 0; p < 4; ++p) {
    const int n = p * 8 + r;
    Wt[(size_t)(n0 + n) * K + k0 + c] = sT[n][c];
  }
}

// ---------------------------------------------------------------------------
// GEMM: C[M,N] = A[M,K](bf16) @ Bt[N,K]^T(bf16) + bias[N], then columns
// [0, scaleCols) are multiplied by oscale (folds 1/sqrt(Dh) into Q).
// Block tile 128x128, K-step 32, 256 threads = 8 waves, wave tile 64x32.
// Double-buffered async LDS staging: copies for step k+1 overlap WMMAs of k.
// ---------------------------------------------------------------------------
#define LDK 40  // padded LDS K-stride (elements); 80B rows, 16B aligned

template <typename OutT>
__global__ __launch_bounds__(256)
void gemm_bf16(const __bf16* __restrict__ A, const __bf16* __restrict__ Bt,
               const float* __restrict__ bias, OutT* __restrict__ C,
               int M, int N, int K, int scaleCols, float oscale) {
  __shared__ __attribute__((aligned(16))) __bf16 sA[2][128 * LDK];  // [m][k]
  __shared__ __attribute__((aligned(16))) __bf16 sB[2][128 * LDK];  // [n][k]

  const int t    = threadIdx.x;
  const int lane = t & 31;
  const int wid  = t >> 5;
  const int wm   = wid >> 2;
  const int wn   = wid & 3;
  const int bm   = blockIdx.y * 128;
  const int bn   = blockIdx.x * 128;
  const int l16  = lane & 15;
  const int half = (lane < 16) ? 0 : 8;
  const int koff = (lane < 16) ? 0 : 16;

  auto stage = [&](int buf, int k0) {
#pragma unroll
    for (int p = 0; p < 2; ++p) {
      const int jj  = t + p * 256;
      const int row = jj >> 2;
      const int seg = (jj & 3) * 8;
      cp16(A  + (size_t)(bm + row) * K + k0 + seg, &sA[buf][row * LDK + seg]);
      cp16(Bt + (size_t)(bn + row) * K + k0 + seg, &sB[buf][row * LDK + seg]);
    }
  };

  v8f acc[4][2];
#pragma unroll
  for (int i = 0; i < 4; ++i)
#pragma unroll
    for (int j = 0; j < 2; ++j) acc[i][j] = (v8f)0.0f;

  stage(0, 0);
  cp_wait();
  __syncthreads();

  const int nk = K / 32;
  for (int ks = 0; ks < nk; ++ks) {
    const int cur = ks & 1;
    if (ks + 1 < nk) stage(cur ^ 1, (ks + 1) * 32);  // overlap with WMMAs

    v16bf af[4], bf[2];
#pragma unroll
    for (int mt = 0; mt < 4; ++mt)
      af[mt] = ld_fragA(&sA[cur][(wm * 64 + mt * 16 + l16) * LDK], half);
#pragma unroll
    for (int nt = 0; nt < 2; ++nt)
      bf[nt] = ld_fragB(&sB[cur][(wn * 32 + nt * 16 + l16) * LDK], koff);
#pragma unroll
    for (int mt = 0; mt < 4; ++mt)
#pragma unroll
      for (int nt = 0; nt < 2; ++nt)
        acc[mt][nt] = WMMA_BF16(af[mt], bf[nt], acc[mt][nt]);

    cp_wait();
    __syncthreads();
  }

#pragma unroll
  for (int mt = 0; mt < 4; ++mt) {
#pragma unroll
    for (int nt = 0; nt < 2; ++nt) {
      const int col = bn + wn * 32 + nt * 16 + l16;
      const float bv = bias[col];
      const float f  = (col < scaleCols) ? oscale : 1.0f;
#pragma unroll
      for (int r = 0; r < 8; ++r) {
        const int row = bm + wm * 64 + mt * 16 + half + r;
        C[(size_t)row * N + col] = (OutT)((acc[mt][nt][r] + bv) * f);
      }
    }
  }
}

// ---------------------------------------------------------------------------
// Flash attention (causal), S^T formulation.  Grid: (C/128, B*H).
// 256 threads = 8 waves; wave w owns query rows [qb*128+16w, +16).
// qkv layout [B,C,3,H,64] bf16 with Q pre-scaled by 1/sqrt(Dh).
// Lane = q column; keys/dh live in VGPRs.  Double-buffered K/V staging.
// ---------------------------------------------------------------------------
#define LDH 72  // padded LDS stride; 144B rows, 16B aligned
#define LDP 40

__global__ __launch_bounds__(256)
void flash_attn(const __bf16* __restrict__ qkv, __bf16* __restrict__ out,
                int Bb, int Cc, int Hh) {
  __shared__ __attribute__((aligned(16))) __bf16 sK[2][64 * LDH];  // [key][dh]
  __shared__ __attribute__((aligned(16))) __bf16 sV[2][64 * LDH];  // [dh][key]
  __shared__ __attribute__((aligned(16))) __bf16 sP[8][16 * LDP];  // P^T per wave

  const int qb   = blockIdx.x;
  const int bh   = blockIdx.y;
  const int b    = bh / Hh;
  const int h    = bh % Hh;
  const int t    = threadIdx.x;
  const int lane = t & 31;
  const int wid  = t >> 5;
  const int l16  = lane & 15;
  const int half = (lane < 16) ? 0 : 8;
  const int koff = (lane < 16) ? 0 : 16;

  const int D3 = 3 * 1024;
  const __bf16* base = qkv + (size_t)b * Cc * D3 + h * 64;

  const int q0    = qb * 128 + wid * 16;
  const int qlane = q0 + l16;
  v16bf qf[2];
  {
    const __bf16* qp = base + (size_t)qlane * D3;  // s = 0 (Q, pre-scaled)
#pragma unroll
    for (int kk = 0; kk < 2; ++kk) qf[kk] = ld_fragB(qp + kk * 32, koff);
  }

  float m_i = -1e30f, l_i = 0.0f;
  v8f o[4];  // O^T: vgpr r -> dh = tt*16 + r + half, lane -> q
#pragma unroll
  for (int tt = 0; tt < 4; ++tt) o[tt] = (v8f)0.0f;

  __bf16* sPw = &sP[wid][0];

  const int nblk = 2 * qb + 2;
  const int key  = t >> 3;        // 0..31
  const int dh   = (t & 7) * 8;   // 0..56

  // ---- prologue: stage block 0 ----
#pragma unroll
  for (int p = 0; p < 2; ++p) {
    const int kkey = p * 32 + key;
    const size_t roff = (size_t)kkey * D3;
    cp16(base + roff + 1024 + dh, &sK[0][kkey * LDH + dh]);   // K (async)
    v8bf vv = *(const v8bf*)(base + roff + 2048 + dh);        // V
#pragma unroll
    for (int i = 0; i < 8; ++i) sV[0][(dh + i) * LDH + kkey] = vv[i];
  }
  cp_wait();
  __syncthreads();

  for (int jb = 0; jb < nblk; ++jb) {
    const int  cur = jb & 1;
    const int  nxt = cur ^ 1;
    const bool pre = (jb + 1 < nblk);

    // ---- issue next block: K async into sK[nxt], V into VGPRs ----
    v8bf vnext[2];
    if (pre) {
#pragma unroll
      for (int p = 0; p < 2; ++p) {
        const int kkey = p * 32 + key;
        const size_t roff = (size_t)((jb + 1) * 64 + kkey) * D3;
        cp16(base + roff + 1024 + dh, &sK[nxt][kkey * LDH + dh]);
        vnext[p] = *(const v8bf*)(base + roff + 2048 + dh);
      }
    }

    for (int ch = 0; ch < 2; ++ch) {
      const int kbase = jb * 64 + ch * 32;
      if (kbase > q0 + 15) continue;  // fully masked for this wave

      // ---- S^T = K_tile @ Q^T : lane = q, vgprs = keys ----
      v8f s0 = (v8f)0.0f, s1 = (v8f)0.0f;
#pragma unroll
      for (int kk = 0; kk < 2; ++kk) {
        v16bf kf0 = ld_fragA(&sK[cur][(ch * 32 + 0  + l16) * LDH + kk * 32], half);
        v16bf kf1 = ld_fragA(&sK[cur][(ch * 32 + 16 + l16) * LDH + kk * 32], half);
        s0 = WMMA_BF16(kf0, qf[kk], s0);
        s1 = WMMA_BF16(kf1, qf[kk], s1);
      }

      // ---- causal mask only on diagonal chunks (Q pre-scaled) ----
      if (kbase + 31 > q0) {
        const int k0b = kbase + half;
#pragma unroll
        for (int e = 0; e < 8; ++e) {
          s0[e] = (k0b + e      <= qlane) ? s0[e] : -1e30f;
          s1[e] = (k0b + 16 + e <= qlane) ? s1[e] : -1e30f;
        }
      }

      // ---- online softmax: in-lane reductions + one xor-16 ----
      float mx = fmaxf(s0[0], s1[0]);
#pragma unroll
      for (int e = 1; e < 8; ++e) mx = fmaxf(mx, fmaxf(s0[e], s1[e]));
      mx = fmaxf(mx, __shfl_xor(mx, 16, 32));
      const float mnew = fmaxf(m_i, mx);
      const float corr = __expf(m_i - mnew);
      m_i = mnew;
#pragma unroll
      for (int e = 0; e < 8; ++e) {
        s0[e] = __expf(s0[e] - mnew);
        s1[e] = __expf(s1[e] - mnew);
      }
      float ps = s0[0] + s1[0];
#pragma unroll
      for (int e = 1; e < 8; ++e) ps += s0[e] + s1[e];
      ps += __shfl_xor(ps, 16, 32);
      l_i = l_i * corr + ps;
#pragma unroll
      for (int tt = 0; tt < 4; ++tt)
#pragma unroll
        for (int r = 0; r < 8; ++r) o[tt][r] *= corr;

      // ---- P^T to wave-private LDS row (packed b32 stores) ----
#pragma unroll
      for (int r = 0; r < 8; r += 2) {
        v2bf p0; p0[0] = (__bf16)s0[r]; p0[1] = (__bf16)s0[r + 1];
        v2bf p1; p1[0] = (__bf16)s1[r]; p1[1] = (__bf16)s1[r + 1];
        *(v2bf*)&sPw[l16 * LDP + half + r]      = p0;
        *(v2bf*)&sPw[l16 * LDP + 16 + half + r] = p1;
      }
      __builtin_amdgcn_wave_barrier();  // same-wave DS ops are in-order
      v16bf pf = ld_fragB(&sPw[l16 * LDP], koff);

      // ---- O^T += V^T_tile @ P^T ----
#pragma unroll
      for (int tt = 0; tt < 4; ++tt) {
        v16bf vf = ld_fragA(&sV[cur][(tt * 16 + l16) * LDH + ch * 32], half);
        o[tt] = WMMA_BF16(vf, pf, o[tt]);
      }
    }

    // ---- store next V (other buffer; nobody reads it before the barrier) ----
    if (pre) {
#pragma unroll
      for (int p = 0; p < 2; ++p) {
        const int kkey = p * 32 + key;
#pragma unroll
        for (int i = 0; i < 8; ++i) sV[nxt][(dh + i) * LDH + kkey] = vnext[p][i];
      }
    }
    cp_wait();
    __syncthreads();
  }

  // ---- finalize: bf16 out[b, q, h*64 + dh]; contiguous 16B per lane ----
  const float inv = 1.0f / l_i;
  __bf16* orow = out + ((size_t)b * Cc + qlane) * 1024 + h * 64 + half;
#pragma unroll
  for (int tt = 0; tt < 4; ++tt) {
    v8bf ov;
#pragma unroll
    for (int r = 0; r < 8; ++r) ov[r] = (__bf16)(o[tt][r] * inv);
    *(v8bf*)(orow + tt * 16) = ov;
  }
}

// ---------------------------------------------------------------------------
extern "C" void kernel_launch(void* const* d_in, const int* in_sizes, int n_in,
                              void* d_out, int out_size, void* d_ws, size_t ws_size,
                              hipStream_t stream) {
  const float* x     = (const float*)d_in[0];
  const float* W_qkv = (const float*)d_in[1];
  const float* b_qkv = (const float*)d_in[2];
  const float* W_o   = (const float*)d_in[3];
  const float* b_o   = (const float*)d_in[4];
  float* out = (float*)d_out;

  const int Bb = 4, Cc = 2048, D = 1024, Hh = 16;
  const int M = Bb * Cc;  // 8192

  char* ws = (char*)d_ws;
  __bf16* qkv   = (__bf16*)ws;  ws += (size_t)M * 3 * D * 2;  // 48 MB
  __bf16* attn  = (__bf16*)ws;  ws += (size_t)M * D * 2;      // 16 MB
  __bf16* x_bf  = (__bf16*)ws;  ws += (size_t)M * D * 2;      // 16 MB
  __bf16* Wqkvt = (__bf16*)ws;  ws += (size_t)D * 3 * D * 2;  //  6 MB
  __bf16* Wot   = (__bf16*)ws;                                //  2 MB

  dim3 blk(256);
  cvt_f32_bf16<<<dim3((M * D / 4 + 255) / 256), blk, 0, stream>>>(x, x_bf, M * D / 4);
  transpose_cvt<<<dim3(3 * D / 32, D / 32), blk, 0, stream>>>(W_qkv, Wqkvt, D, 3 * D);
  transpose_cvt<<<dim3(D / 32, D / 32), blk, 0, stream>>>(W_o, Wot, D, D);

  // qkv = x @ W_qkv + b_qkv; Q columns (0..1023) pre-scaled by 1/sqrt(64)
  gemm_bf16<__bf16><<<dim3((3 * D) / 128, M / 128), blk, 0, stream>>>(
      x_bf, Wqkvt, b_qkv, qkv, M, 3 * D, D, D, 0.125f);
  flash_attn<<<dim3(Cc / 128, Bb * Hh), blk, 0, stream>>>(qkv, attn, Bb, Cc, Hh);
  gemm_bf16<float><<<dim3(D / 128, M / 128), blk, 0, stream>>>(
      attn, Wot, b_o, out, M, D, D, 0, 1.0f);
}
